// EpisodeMultiheadAttentionBlock_90417651516093
// MI455X (gfx1250) — matrix operand order repaired and checked by reading, not compiled, not run on hardware
//
#include <hip/hip_runtime.h>
#include <hip/hip_bf16.h>

// Sizes fixed by the reference
#define EMB   1024
#define NH    16
#define DH    64
#define BATCH 8
#define LKK   1024
#define LQQ   1024
#define MROWS (BATCH*LKK)   // 8192
#define N_QKV (3*EMB)       // 3072

typedef __attribute__((ext_vector_type(16))) __bf16        v16bf;
typedef __attribute__((ext_vector_type(8)))  float         v8f;
typedef __attribute__((ext_vector_type(8)))  unsigned short ushort8;

union FragU { v16bf bf; ushort8 u[2]; };

__device__ __forceinline__ v8f wmma_bf16(v16bf a, v16bf b, v8f c) {
    return __builtin_amdgcn_wmma_f32_16x16x32_bf16(
        /*neg_a=*/false, a, /*neg_b=*/false, b,
        /*c_mod=*/(short)0, c, /*reuse_a=*/false, /*reuse_b=*/false);
}

// A-fragment: 16x32 bf16 tile, row-major, `ld` element stride.
// lane<16 : row m=lane,  K = {kb..kb+7, kb+16..kb+23}, kb=0
// lane>=16: row m=l&15,  kb=8
__device__ __forceinline__ v16bf load_frag_a(const __hip_bfloat16* tile, int ld) {
    const int l  = threadIdx.x & 31;
    const int m  = l & 15;
    const int kb = (l >> 4) << 3;           // 0 or 8
    FragU f;
    f.u[0] = *(const ushort8*)(tile + m * ld + kb);
    f.u[1] = *(const ushort8*)(tile + m * ld + kb + 16);
    return f.bf;
}

// B-fragment: 32x16 bf16, loaded from the row-major [N][K] view (B transposed).
// lane holds its column's K = {kb..kb+7, kb+16..kb+23}: lanes<16 kb=0, lanes>=16 kb=8.
__device__ __forceinline__ v16bf load_frag_b(const __hip_bfloat16* tileNK, int ld) {
    const int l  = threadIdx.x & 31;
    const int n  = l & 15;
    const int kb = (l >> 4) << 3;           // 0 or 8
    const ushort8* p = (const ushort8*)(tileNK + n * ld + kb);
    FragU f;
    f.u[0] = p[0];
    f.u[1] = p[2];
    return f.bf;
}

// ---------------------------------------------------------------- utilities
__global__ void k_f32_to_bf16(const float* __restrict__ src,
                              __hip_bfloat16* __restrict__ dst, int n) {
    int i = blockIdx.x * 256 + threadIdx.x;
    if (i < n) dst[i] = __float2bfloat16(src[i]);
}

__global__ void k_zero_f32(float* __restrict__ p, int n) {
    int i = blockIdx.x * 256 + threadIdx.x;
    if (i < n) p[i] = 0.0f;
}

// ---------------------------------------------------------------- QKV GEMM
// Y[M=8192, N=3072] = Xb @ Wb^T + bias ; scatter into Q, K (row-major [bh][L][64])
// and V transposed ([bh][64][Lk]). Block = 4 waves, 64x64 tile per block.
__global__ void __launch_bounds__(128)
k_qkv_gemm(const __hip_bfloat16* __restrict__ Xb,     // [8192][1024]
           const __hip_bfloat16* __restrict__ Wb,     // [3072][1024]
           const float* __restrict__ bias,            // [3072]
           __hip_bfloat16* __restrict__ Qb,           // [B*H][LQ][64]
           __hip_bfloat16* __restrict__ Kb,           // [B*H][LK][64]
           __hip_bfloat16* __restrict__ VTb)          // [B*H][64][LK]
{
    const int w  = threadIdx.x >> 5;
    const int l  = threadIdx.x & 31;
    const int mt = blockIdx.x * 64 + w * 16;
    const int nt = blockIdx.y * 64;

    v8f zf = {};
    v8f c[4];
#pragma unroll
    for (int j = 0; j < 4; ++j) c[j] = zf;

    const __hip_bfloat16* a_row = Xb + (size_t)mt * EMB;
    for (int kk = 0; kk < EMB; kk += 32) {
        __builtin_prefetch(a_row + kk + 128, 0, 1);
        v16bf a = load_frag_a(a_row + kk, EMB);
#pragma unroll
        for (int j = 0; j < 4; ++j) {
            v16bf b = load_frag_b(Wb + (size_t)(nt + 16 * j) * EMB + kk, EMB);
            c[j] = wmma_bf16(a, b, c[j]);
        }
    }

    const int nl = l & 15;
    const int rl = (l >> 4) * 8;
#pragma unroll
    for (int j = 0; j < 4; ++j) {
        const int col   = nt + 16 * j + nl;
        const int which = col >> 10;        // 0:q 1:k 2:v
        const int e     = col & (EMB - 1);
        const int hh    = e >> 6;
        const int dd    = e & 63;
        const float bc  = bias[col];
#pragma unroll
        for (int i = 0; i < 8; ++i) {
            const int row = mt + rl + i;
            const int bb  = row >> 10;
            const int lk  = row & (LKK - 1);
            const int bh  = bb * NH + hh;
            const __hip_bfloat16 hv = __float2bfloat16(c[j][i] + bc);
            if (which == 0)      Qb[((size_t)bh * LQQ + lk) * DH + dd] = hv;
            else if (which == 1) Kb[((size_t)bh * LKK + lk) * DH + dd] = hv;
            else                 VTb[((size_t)bh * DH + dd) * LKK + lk] = hv;
        }
    }
}

// ---------------------------------------------------------------- attention
// Grid: (LQ/64, B*H). Block = 4 waves; wave w owns 16 query rows.
// Two-pass softmax (online max/sum, then normalized P), WMMA for S and P@V,
// P staged through per-wave LDS tile to re-layout C-frag -> A-frag.
__global__ void __launch_bounds__(128)
k_attn(const __hip_bfloat16* __restrict__ Qb,
       const __hip_bfloat16* __restrict__ Kb,
       const __hip_bfloat16* __restrict__ VTb,
       const unsigned char* __restrict__ kpm,   // [B][LK] bool
       float* __restrict__ attnw,               // [B][LQ][LK] (head mean, atomically acc.)
       __hip_bfloat16* __restrict__ ctxb)       // [B][LQ][E]
{
    __shared__ __align__(16) __hip_bfloat16 pbuf[4][16 * 32];

    const int w  = threadIdx.x >> 5;
    const int l  = threadIdx.x & 31;
    const int bh = blockIdx.y;
    const int bb = bh >> 4;
    const int hh = bh & 15;
    const int qr0 = blockIdx.x * 64 + w * 16;
    const int nl = l & 15;
    const int rl = (l >> 4) * 8;

    const __hip_bfloat16* Qbh = Qb  + (size_t)bh * LQQ * DH;
    const __hip_bfloat16* Kbh = Kb  + (size_t)bh * LKK * DH;
    const __hip_bfloat16* Vbh = VTb + (size_t)bh * DH  * LKK;

    v16bf aq0 = load_frag_a(Qbh + (size_t)qr0 * DH,      DH);
    v16bf aq1 = load_frag_a(Qbh + (size_t)qr0 * DH + 32, DH);

    const float scale = 0.125f;               // 1/sqrt(64)
    float mrow[8], lrow[8];
#pragma unroll
    for (int i = 0; i < 8; ++i) { mrow[i] = -3.0e38f; lrow[i] = 0.0f; }

    // ---- pass 1: online row max / sum over all key tiles
    for (int kt = 0; kt < LKK / 16; ++kt) {
        const int k0 = kt * 16;
        v8f s = {};
        v16bf bk0 = load_frag_b(Kbh + (size_t)k0 * DH,      DH);
        v16bf bk1 = load_frag_b(Kbh + (size_t)k0 * DH + 32, DH);
        s = wmma_bf16(aq0, bk0, s);
        s = wmma_bf16(aq1, bk1, s);
        const int  kg  = k0 + nl;
        const bool pad = kpm[bb * LKK + kg] != 0;
#pragma unroll
        for (int i = 0; i < 8; ++i) {
            const int qg = qr0 + rl + i;
            float sv = s[i] * scale;
            if (kg != qg && ((kg > qg) || pad)) sv = -3.0e38f;  // causal|pad, diag kept
            float tm = sv;
            tm = fmaxf(tm, __shfl_xor(tm, 1, 32));
            tm = fmaxf(tm, __shfl_xor(tm, 2, 32));
            tm = fmaxf(tm, __shfl_xor(tm, 4, 32));
            tm = fmaxf(tm, __shfl_xor(tm, 8, 32));
            const float mnew = fmaxf(mrow[i], tm);
            float p = __expf(sv - mnew);
            p += __shfl_xor(p, 1, 32);
            p += __shfl_xor(p, 2, 32);
            p += __shfl_xor(p, 4, 32);
            p += __shfl_xor(p, 8, 32);
            lrow[i] = lrow[i] * __expf(mrow[i] - mnew) + p;
            mrow[i] = mnew;
        }
    }

    float linv[8];
#pragma unroll
    for (int i = 0; i < 8; ++i) linv[i] = 1.0f / lrow[i];

    v8f zf = {};
    v8f cacc[4];
#pragma unroll
    for (int dt = 0; dt < 4; ++dt) cacc[dt] = zf;

    const float invH = 1.0f / (float)NH;
    __hip_bfloat16* pb = &pbuf[w][0];

    // ---- pass 2: normalized P -> attn mean (atomic) + ctx = P @ V
    for (int t2 = 0; t2 < LKK / 32; ++t2) {
#pragma unroll
        for (int sub = 0; sub < 2; ++sub) {
            const int k0 = t2 * 32 + sub * 16;
            v8f s = {};
            v16bf bk0 = load_frag_b(Kbh + (size_t)k0 * DH,      DH);
            v16bf bk1 = load_frag_b(Kbh + (size_t)k0 * DH + 32, DH);
            s = wmma_bf16(aq0, bk0, s);
            s = wmma_bf16(aq1, bk1, s);
            const int  kg  = k0 + nl;
            const bool pad = kpm[bb * LKK + kg] != 0;
#pragma unroll
            for (int i = 0; i < 8; ++i) {
                const int qg = qr0 + rl + i;
                float sv = s[i] * scale;
                if (kg != qg && ((kg > qg) || pad)) sv = -3.0e38f;
                const float p = __expf(sv - mrow[i]) * linv[i];
                unsafeAtomicAdd(&attnw[((size_t)bb * LQQ + qg) * LKK + kg], p * invH);
                pb[(rl + i) * 32 + sub * 16 + nl] = __float2bfloat16(p);
            }
        }
        // same-wave LDS RAW is in-order; compiler inserts s_wait_dscnt
        v16bf pa = load_frag_a(pb, 32);
#pragma unroll
        for (int dt = 0; dt < 4; ++dt) {
            v16bf bv = load_frag_b(Vbh + (size_t)(dt * 16) * LKK + t2 * 32, LKK);
            cacc[dt] = wmma_bf16(pa, bv, cacc[dt]);
        }
    }

#pragma unroll
    for (int dt = 0; dt < 4; ++dt) {
#pragma unroll
        for (int i = 0; i < 8; ++i) {
            const int qg = qr0 + rl + i;
            ctxb[((size_t)bb * LQQ + qg) * EMB + hh * DH + dt * 16 + nl] =
                __float2bfloat16(cacc[dt][i]);
        }
    }
}

// ---------------------------------------------------------------- out proj
__global__ void __launch_bounds__(128)
k_outproj(const __hip_bfloat16* __restrict__ Cb,    // ctx [8192][1024]
          const __hip_bfloat16* __restrict__ Wb,    // out_w bf16 [1024][1024]
          const float* __restrict__ bias,           // [1024]
          float* __restrict__ opre)                 // [8192][1024] fp32
{
    const int w  = threadIdx.x >> 5;
    const int l  = threadIdx.x & 31;
    const int mt = blockIdx.x * 64 + w * 16;
    const int nt = blockIdx.y * 64;

    v8f zf = {};
    v8f c[4];
#pragma unroll
    for (int j = 0; j < 4; ++j) c[j] = zf;

    const __hip_bfloat16* a_row = Cb + (size_t)mt * EMB;
    for (int kk = 0; kk < EMB; kk += 32) {
        __builtin_prefetch(a_row + kk + 128, 0, 1);
        v16bf a = load_frag_a(a_row + kk, EMB);
#pragma unroll
        for (int j = 0; j < 4; ++j) {
            v16bf b = load_frag_b(Wb + (size_t)(nt + 16 * j) * EMB + kk, EMB);
            c[j] = wmma_bf16(a, b, c[j]);
        }
    }

    const int nl = l & 15;
    const int rl = (l >> 4) * 8;
#pragma unroll
    for (int j = 0; j < 4; ++j) {
        const int col  = nt + 16 * j + nl;
        const float bc = bias[col];
#pragma unroll
        for (int i = 0; i < 8; ++i) {
            const int row = mt + rl + i;
            opre[(size_t)row * EMB + col] = c[j][i] + bc;
        }
    }
}

// ---------------------------------------------------------------- layernorm + residual
__global__ void __launch_bounds__(256)
k_ln(const float* __restrict__ opre,
     const float* __restrict__ resid,   // query == key (Lq == Lk)
     const float* __restrict__ g, const float* __restrict__ bln,
     float* __restrict__ out)
{
    __shared__ float red[16];
    const int row  = blockIdx.x;
    const int tid  = threadIdx.x;
    const int lane = tid & 31;
    const int wid  = tid >> 5;

    const float* x = opre + (size_t)row * EMB;
    const float4 v = ((const float4*)x)[tid];       // 256*4 = 1024
    float sum = v.x + v.y + v.z + v.w;
    float sq  = v.x * v.x + v.y * v.y + v.z * v.z + v.w * v.w;
#pragma unroll
    for (int m = 16; m >= 1; m >>= 1) {
        sum += __shfl_xor(sum, m, 32);
        sq  += __shfl_xor(sq,  m, 32);
    }
    if (lane == 0) { red[wid] = sum; red[8 + wid] = sq; }
    __syncthreads();
    float tot = 0.0f, tot2 = 0.0f;
#pragma unroll
    for (int i = 0; i < 8; ++i) { tot += red[i]; tot2 += red[8 + i]; }
    const float mu   = tot * (1.0f / EMB);
    const float var  = tot2 * (1.0f / EMB) - mu * mu;
    const float rstd = rsqrtf(var + 1e-5f);

    const int e0 = tid * 4;
#pragma unroll
    for (int cidx = 0; cidx < 4; ++cidx) {
        const int e = e0 + cidx;
        const float xv = (cidx == 0) ? v.x : (cidx == 1) ? v.y : (cidx == 2) ? v.z : v.w;
        out[(size_t)row * EMB + e] =
            (xv - mu) * rstd * g[e] + bln[e] + resid[(size_t)row * EMB + e];
    }
}

// ---------------------------------------------------------------- launch
extern "C" void kernel_launch(void* const* d_in, const int* in_sizes, int n_in,
                              void* d_out, int out_size, void* d_ws, size_t ws_size,
                              hipStream_t stream) {
    const float*         key      = (const float*)d_in[0];
    // d_in[1] = query_length (== 1024, compile-time constant here)
    const unsigned char* kpm      = (const unsigned char*)d_in[2];
    const float*         in_w     = (const float*)d_in[3];
    const float*         in_b     = (const float*)d_in[4];
    const float*         out_w    = (const float*)d_in[5];
    const float*         out_b    = (const float*)d_in[6];
    const float*         ln_g     = (const float*)d_in[7];
    const float*         ln_b     = (const float*)d_in[8];

    float* out   = (float*)d_out;
    float* attnw = out + (size_t)MROWS * EMB;       // second tuple element

    char* ws = (char*)d_ws;
    __hip_bfloat16* Xb  = (__hip_bfloat16*)ws; ws += (size_t)MROWS * EMB * 2;
    __hip_bfloat16* Wb  = (__hip_bfloat16*)ws; ws += (size_t)N_QKV * EMB * 2;
    __hip_bfloat16* OWb = (__hip_bfloat16*)ws; ws += (size_t)EMB * EMB * 2;
    __hip_bfloat16* Qb  = (__hip_bfloat16*)ws; ws += (size_t)BATCH * NH * LQQ * DH * 2;
    __hip_bfloat16* Kb  = (__hip_bfloat16*)ws; ws += (size_t)BATCH * NH * LKK * DH * 2;
    __hip_bfloat16* VTb = (__hip_bfloat16*)ws; ws += (size_t)BATCH * NH * DH * LKK * 2;
    __hip_bfloat16* Cb  = (__hip_bfloat16*)ws; ws += (size_t)MROWS * EMB * 2;
    float*          opre = (float*)ws;

    const int nKey = MROWS * EMB;        // 8388608
    const int nInW = N_QKV * EMB;        // 3145728
    const int nOW  = EMB * EMB;          // 1048576

    k_f32_to_bf16<<<(nKey + 255) / 256, 256, 0, stream>>>(key,   Xb,  nKey);
    k_f32_to_bf16<<<(nInW + 255) / 256, 256, 0, stream>>>(in_w,  Wb,  nInW);
    k_f32_to_bf16<<<(nOW  + 255) / 256, 256, 0, stream>>>(out_w, OWb, nOW);
    k_zero_f32   <<<(nKey + 255) / 256, 256, 0, stream>>>(attnw, nKey);

    k_qkv_gemm<<<dim3(MROWS / 64, N_QKV / 64), 128, 0, stream>>>(
        Xb, Wb, in_b, Qb, Kb, VTb);

    k_attn<<<dim3(LQQ / 64, BATCH * NH), 128, 0, stream>>>(
        Qb, Kb, VTb, kpm, attnw, Cb);

    k_outproj<<<dim3(MROWS / 64, EMB / 64), 128, 0, stream>>>(
        Cb, OWb, out_b, opre);

    k_ln<<<MROWS, 256, 0, stream>>>(opre, key, ln_g, ln_b, out);
}
